// EncodeProcessDecode_32109175505238
// MI455X (gfx1250) — matrix operand order repaired and checked by reading, not compiled, not run on hardware
//
#include <hip/hip_runtime.h>
#include <hip/hip_bf16.h>

typedef __attribute__((ext_vector_type(16))) _Float16 v16h;
typedef __attribute__((ext_vector_type(8)))  _Float16 v8h;
typedef __attribute__((ext_vector_type(8)))  float    v8f;

#define HD   128     // hidden width
#define LDA  136     // LDS row stride (halves): 272B rows -> 16B-aligned fragments
#define ALD  48      // encoder input tile stride (96B rows -> 16B aligned)

// ---------------------------------------------------------------------------
// WMMA fragment layouts per CDNA5 ISA 7.12.2 (wave32):
// A: 16x32 f16. lane L: M=L&15; hi=L>>4. elem e -> K = kt*32 + (e&7) + ((e>>3)<<4) + hi*8
// B: 32x16 f16. lane L: N=L&15.  elem e -> K = kt*32 + (e&7) + ((e>>3)<<4) + hi*8
// C/D: 16x16 f32. vgpr r: M = r + 8*hi; N = lane&15
// B is PRE-PACKED fragment-major: dst[((kt*8+nt)*32+lane)*16 + e]
// ---------------------------------------------------------------------------
__device__ __forceinline__ v16h cat8(v8h lo, v8h hi) {
  return __builtin_shufflevector(lo, hi, 0, 1, 2, 3, 4, 5, 6, 7,
                                 8, 9, 10, 11, 12, 13, 14, 15);
}

__device__ __forceinline__ v16h load_a_frag(const _Float16* lds, int lda, int row_base, int kt) {
  const int lane = threadIdx.x & 31;
  const int hi   = lane >> 4;
  const int row  = row_base + (lane & 15);
  const _Float16* p = lds + row * lda + kt * 32 + hi * 8;
  v8h lo = *(const v8h*)p;          // K +0..7   (16B aligned)
  v8h hh = *(const v8h*)(p + 16);   // K +16..23 (16B aligned)
  return cat8(lo, hh);
}

__device__ __forceinline__ v16h load_b_packed(const _Float16* __restrict__ Wp, int kt, int nt) {
  const int lane = threadIdx.x & 31;
  const _Float16* p = Wp + (((size_t)((kt * 8 + nt) * 32 + lane)) << 4);
  v8h lo = *(const v8h*)p;          // 32B/lane, fully coalesced
  v8h hh = *(const v8h*)(p + 8);
  return cat8(lo, hh);
}

__device__ __forceinline__ void zero_acc(v8f acc[8]) {
#pragma unroll
  for (int i = 0; i < 8; ++i)
#pragma unroll
    for (int j = 0; j < 8; ++j) acc[i][j] = 0.0f;
}

// one wave computes TWO 16x128 strips (rows wr0..+15 and wr1..+15), sharing B frags.
// B fragments are double-buffered so global loads overlap the WMMA pipe.
template <int KT>
__device__ __forceinline__ void gemm_strip2(const _Float16* Alds, int lda, int wr0, int wr1,
                                            const _Float16* __restrict__ Wp,
                                            v8f acc0[8], v8f acc1[8]) {
#pragma unroll
  for (int kt = 0; kt < KT; ++kt) {
    v16h a0 = load_a_frag(Alds, lda, wr0, kt);
    v16h a1 = load_a_frag(Alds, lda, wr1, kt);
    v16h b  = load_b_packed(Wp, kt, 0);
#pragma unroll
    for (int nt = 0; nt < 8; ++nt) {
      v16h bn;
      if (nt < 7) bn = load_b_packed(Wp, kt, nt + 1);
      acc0[nt] = __builtin_amdgcn_wmma_f32_16x16x32_f16(
          false, a0, false, b, (short)0, acc0[nt], false, false);
      acc1[nt] = __builtin_amdgcn_wmma_f32_16x16x32_f16(
          false, a1, false, b, (short)0, acc1[nt], false, false);
      if (nt < 7) b = bn;
    }
  }
}

// store strip (+bias, relu) into LDS f16 (next GEMM's A operand)
__device__ __forceinline__ void store_h_lds(_Float16* Hs, int lda, int row_base,
                                            const v8f acc[8], const float* __restrict__ bias) {
  const int lane = threadIdx.x & 31;
  const int hi   = lane >> 4;
#pragma unroll
  for (int nt = 0; nt < 8; ++nt) {
    const int col = nt * 16 + (lane & 15);
    const float bb = bias[col];
#pragma unroll
    for (int r = 0; r < 8; ++r) {
      float v = fmaxf(acc[nt][r] + bb, 0.0f);
      Hs[(row_base + r + hi * 8) * lda + col] = (_Float16)v;
    }
  }
}

__device__ __forceinline__ void store_f32_global(float* __restrict__ out, int row0, int M,
                                                 int row_base, const v8f acc[8],
                                                 const float* __restrict__ bias) {
  const int lane = threadIdx.x & 31;
  const int hi   = lane >> 4;
#pragma unroll
  for (int nt = 0; nt < 8; ++nt) {
    const int col = nt * 16 + (lane & 15);
    const float bb = bias ? bias[col] : 0.0f;
#pragma unroll
    for (int r = 0; r < 8; ++r) {
      int row = row0 + row_base + r + hi * 8;
      if (row < M) out[(size_t)row * HD + col] = acc[nt][r] + bb;
    }
  }
}

__device__ __forceinline__ void store_f16_global(_Float16* __restrict__ out, int row0, int M,
                                                 int row_base, const v8f acc[8],
                                                 const float* __restrict__ bias) {
  const int lane = threadIdx.x & 31;
  const int hi   = lane >> 4;
#pragma unroll
  for (int nt = 0; nt < 8; ++nt) {
    const int col = nt * 16 + (lane & 15);
    const float bb = bias[col];
#pragma unroll
    for (int r = 0; r < 8; ++r) {
      int row = row0 + row_base + r + hi * 8;
      if (row < M) out[(size_t)row * HD + col] = (_Float16)(acc[nt][r] + bb);
    }
  }
}

// ---------------------------------------------------------------------------
// pack f32 weight [K x 128] -> f16 fragment-major layout (zero-pad K to kt*32)
// ---------------------------------------------------------------------------
__global__ void k_pack_b(_Float16* __restrict__ dst, const float* __restrict__ src,
                         int ktiles, int K) {
  int idx = blockIdx.x * blockDim.x + threadIdx.x;
  int total = ktiles * 4096;  // ktiles*8nt*32lane*16e
  if (idx >= total) return;
  int e    = idx & 15;
  int lane = (idx >> 4) & 31;
  int nt   = (idx >> 9) & 7;
  int kt   = idx >> 12;
  int k    = kt * 32 + (e & 7) + ((e >> 3) << 4) + (lane >> 4) * 8;
  int col  = nt * 16 + (lane & 15);
  float v  = (k < K) ? src[(size_t)k * HD + col] : 0.0f;
  dst[idx] = (_Float16)v;
}

__global__ void k_f2h(_Float16* __restrict__ dst, const float* __restrict__ src, int n) {
  int i = blockIdx.x * blockDim.x + threadIdx.x;
  if (i < n) dst[i] = (_Float16)src[i];
}

// ---------------------------------------------------------------------------
// 2-layer encoder MLP: out = (relu(X@W0+b0))@W1 + b1.  Din in {16,32}.
// 64 rows/block, 64 threads (2 waves x 2 strips).
// ---------------------------------------------------------------------------
__global__ void __launch_bounds__(64, 1)
k_encode(const float* __restrict__ X, int Din, int M,
         const _Float16* __restrict__ W0p, const float* __restrict__ b0,
         const _Float16* __restrict__ W1p, const float* __restrict__ b1,
         float* __restrict__ outF, _Float16* __restrict__ outH) {
  __shared__ __align__(16) _Float16 A[64 * ALD];
  __shared__ __align__(16) _Float16 Hs[64 * LDA];
  const int r0 = blockIdx.x * 64;
  for (int idx = threadIdx.x; idx < 64 * ALD; idx += 64) {
    int r = idx / ALD, c = idx % ALD;
    int row = r0 + r;
    float v = (c < Din && row < M) ? X[(size_t)row * Din + c] : 0.0f;
    A[idx] = (_Float16)v;
  }
  __syncthreads();
  const int wv = threadIdx.x >> 5;
  const int wr0 = wv * 16, wr1 = 32 + wv * 16;
  v8f acc0[8], acc1[8];
  zero_acc(acc0); zero_acc(acc1);
  gemm_strip2<1>(A, ALD, wr0, wr1, W0p, acc0, acc1);
  store_h_lds(Hs, LDA, wr0, acc0, b0);
  store_h_lds(Hs, LDA, wr1, acc1, b0);
  __syncthreads();
  zero_acc(acc0); zero_acc(acc1);
  gemm_strip2<4>(Hs, LDA, wr0, wr1, W1p, acc0, acc1);
  if (outF) {
    store_f32_global(outF, r0, M, wr0, acc0, b1);
    store_f32_global(outF, r0, M, wr1, acc1, b1);
  } else {
    store_f16_global(outH, r0, M, wr0, acc0, b1);
    store_f16_global(outH, r0, M, wr1, acc1, b1);
  }
}

// ---------------------------------------------------------------------------
// fused message + scatter: msg = [n[snd] | e] @ W_msg ; agg[rcv] += msg
// 64 edges/block, 64 threads.
// ---------------------------------------------------------------------------
__global__ void __launch_bounds__(64, 1)
k_message(const float* __restrict__ n, const _Float16* __restrict__ e16,
          const int* __restrict__ senders, const int* __restrict__ receivers,
          const _Float16* __restrict__ Wms, const _Float16* __restrict__ Wme,
          float* __restrict__ agg, int E) {
  __shared__ __align__(16) _Float16 As[64 * LDA];
  __shared__ __align__(16) _Float16 Ae[64 * LDA];
  const int e0 = blockIdx.x * 64;
  const int t = threadIdx.x;  // 0..63
  for (int it = 0; it < 64; ++it) {
    int edge = e0 + it;
    float s0 = 0.0f, s1 = 0.0f;
    _Float16 ev0 = (_Float16)0.0f, ev1 = (_Float16)0.0f;
    if (edge < E) {
      const float* nr = n + (size_t)senders[edge] * HD;
      s0 = nr[t]; s1 = nr[t + 64];
      const _Float16* er = e16 + (size_t)edge * HD;
      ev0 = er[t]; ev1 = er[t + 64];
    }
    As[it * LDA + t] = (_Float16)s0;
    As[it * LDA + t + 64] = (_Float16)s1;
    Ae[it * LDA + t] = ev0;
    Ae[it * LDA + t + 64] = ev1;
  }
  __syncthreads();
  const int wv = threadIdx.x >> 5;
  const int wr0 = wv * 16, wr1 = 32 + wv * 16;
  v8f acc0[8], acc1[8];
  zero_acc(acc0); zero_acc(acc1);
  gemm_strip2<4>(As, LDA, wr0, wr1, Wms, acc0, acc1);
  gemm_strip2<4>(Ae, LDA, wr0, wr1, Wme, acc0, acc1);
  const int lane = threadIdx.x & 31, hi = lane >> 4;
#pragma unroll
  for (int r = 0; r < 8; ++r) {
    int er0 = e0 + wr0 + r + hi * 8;
    if (er0 < E) {
      float* aggrow = agg + (size_t)receivers[er0] * HD;
#pragma unroll
      for (int nt = 0; nt < 8; ++nt)
        atomicAdd(&aggrow[nt * 16 + (lane & 15)], acc0[nt][r]);
    }
    int er1 = e0 + wr1 + r + hi * 8;
    if (er1 < E) {
      float* aggrow = agg + (size_t)receivers[er1] * HD;
#pragma unroll
      for (int nt = 0; nt < 8; ++nt)
        atomicAdd(&aggrow[nt * 16 + (lane & 15)], acc1[nt][r]);
    }
  }
}

// ---------------------------------------------------------------------------
// node update (pre-LayerNorm): u = n@W_node + MLP([n|agg]); 64 nodes/block
// ---------------------------------------------------------------------------
__global__ void __launch_bounds__(64, 1)
k_node_update(const float* __restrict__ n, const float* __restrict__ agg,
              const _Float16* __restrict__ W0a, const _Float16* __restrict__ W0b,
              const float* __restrict__ b0,
              const _Float16* __restrict__ W1p, const float* __restrict__ b1,
              const _Float16* __restrict__ Wnp,
              float* __restrict__ u, int M) {
  __shared__ __align__(16) _Float16 An[64 * LDA];
  __shared__ __align__(16) _Float16 Aa[64 * LDA];
  __shared__ __align__(16) _Float16 Hs[64 * LDA];
  const int r0 = blockIdx.x * 64;
  const int t = threadIdx.x;
  for (int it = 0; it < 64; ++it) {
    int row = r0 + it;
    float n0 = 0.0f, n1 = 0.0f, a0 = 0.0f, a1 = 0.0f;
    if (row < M) {
      const float* nr = n + (size_t)row * HD;
      const float* ar = agg + (size_t)row * HD;
      n0 = nr[t]; n1 = nr[t + 64];
      a0 = ar[t]; a1 = ar[t + 64];
    }
    An[it * LDA + t] = (_Float16)n0;
    An[it * LDA + t + 64] = (_Float16)n1;
    Aa[it * LDA + t] = (_Float16)a0;
    Aa[it * LDA + t + 64] = (_Float16)a1;
  }
  __syncthreads();
  const int wv = threadIdx.x >> 5;
  const int wr0 = wv * 16, wr1 = 32 + wv * 16;
  v8f acc0[8], acc1[8];
  zero_acc(acc0); zero_acc(acc1);
  gemm_strip2<4>(An, LDA, wr0, wr1, W0a, acc0, acc1);  // node_W0 rows 0..127 (n part)
  gemm_strip2<4>(Aa, LDA, wr0, wr1, W0b, acc0, acc1);  // node_W0 rows 128..255 (agg part)
  store_h_lds(Hs, LDA, wr0, acc0, b0);
  store_h_lds(Hs, LDA, wr1, acc1, b0);
  __syncthreads();
  zero_acc(acc0); zero_acc(acc1);
  gemm_strip2<4>(Hs, LDA, wr0, wr1, W1p, acc0, acc1);  // MLP layer 2
  gemm_strip2<4>(An, LDA, wr0, wr1, Wnp, acc0, acc1);  // residual n @ W_node
  store_f32_global(u, r0, M, wr0, acc0, b1);
  store_f32_global(u, r0, M, wr1, acc1, b1);
}

// ---------------------------------------------------------------------------
// in-place LayerNorm over last dim (128); one wave32 per row
// ---------------------------------------------------------------------------
__global__ void __launch_bounds__(256)
k_layernorm(float* __restrict__ u, const float* __restrict__ g,
            const float* __restrict__ b, int M) {
  int wid = (blockIdx.x * blockDim.x + threadIdx.x) >> 5;
  int lane = threadIdx.x & 31;
  if (wid >= M) return;
  float* row = u + (size_t)wid * HD;
  float v0 = row[lane], v1 = row[lane + 32], v2 = row[lane + 64], v3 = row[lane + 96];
  float s = v0 + v1 + v2 + v3;
  float q = v0 * v0 + v1 * v1 + v2 * v2 + v3 * v3;
#pragma unroll
  for (int m = 16; m >= 1; m >>= 1) {
    s += __shfl_xor(s, m, 32);
    q += __shfl_xor(q, m, 32);
  }
  float mean = s * (1.0f / 128.0f);
  float var  = q * (1.0f / 128.0f) - mean * mean;
  float rstd = rsqrtf(var + 1e-6f);
  row[lane]      = (v0 - mean) * rstd * g[lane]      + b[lane];
  row[lane + 32] = (v1 - mean) * rstd * g[lane + 32] + b[lane + 32];
  row[lane + 64] = (v2 - mean) * rstd * g[lane + 64] + b[lane + 64];
  row[lane + 96] = (v3 - mean) * rstd * g[lane + 96] + b[lane + 96];
}

// ---------------------------------------------------------------------------
// decoder: hidden = relu(n@W0+b0) via WMMA; out = hidden@W1(128x2)+b1 scalar
// ---------------------------------------------------------------------------
__global__ void __launch_bounds__(64, 1)
k_decode(const float* __restrict__ n,
         const _Float16* __restrict__ W0p, const float* __restrict__ b0,
         const _Float16* __restrict__ W1, const float* __restrict__ b1,
         float* __restrict__ out, int M) {
  __shared__ __align__(16) _Float16 A[64 * LDA];
  __shared__ __align__(16) _Float16 Hs[64 * LDA];
  const int r0 = blockIdx.x * 64;
  const int t = threadIdx.x;
  for (int it = 0; it < 64; ++it) {
    int row = r0 + it;
    float v0 = 0.0f, v1 = 0.0f;
    if (row < M) {
      const float* nr = n + (size_t)row * HD;
      v0 = nr[t]; v1 = nr[t + 64];
    }
    A[it * LDA + t] = (_Float16)v0;
    A[it * LDA + t + 64] = (_Float16)v1;
  }
  __syncthreads();
  const int wv = threadIdx.x >> 5;
  const int wr0 = wv * 16, wr1 = 32 + wv * 16;
  v8f acc0[8], acc1[8];
  zero_acc(acc0); zero_acc(acc1);
  gemm_strip2<4>(A, LDA, wr0, wr1, W0p, acc0, acc1);
  store_h_lds(Hs, LDA, wr0, acc0, b0);
  store_h_lds(Hs, LDA, wr1, acc1, b0);
  __syncthreads();
  int grow = r0 + t;  // thread t handles row t, both output cols
  if (grow < M) {
    float s0 = b1[0], s1 = b1[1];
#pragma unroll 8
    for (int k = 0; k < HD; ++k) {
      float h = (float)Hs[t * LDA + k];
      s0 += h * (float)W1[k * 2 + 0];
      s1 += h * (float)W1[k * 2 + 1];
    }
    out[(size_t)grow * 2 + 0] = s0;
    out[(size_t)grow * 2 + 1] = s1;
  }
}

// ---------------------------------------------------------------------------
// host launcher
// ---------------------------------------------------------------------------
extern "C" void kernel_launch(void* const* d_in, const int* in_sizes, int n_in,
                              void* d_out, int out_size, void* d_ws, size_t ws_size,
                              hipStream_t stream) {
  const float* nodes     = (const float*)d_in[0];
  const float* edges     = (const float*)d_in[1];
  const int*   senders   = (const int*)d_in[2];
  const int*   receivers = (const int*)d_in[3];
  const float* enc_n_W0 = (const float*)d_in[4];
  const float* enc_n_b0 = (const float*)d_in[5];
  const float* enc_n_W1 = (const float*)d_in[6];
  const float* enc_n_b1 = (const float*)d_in[7];
  const float* enc_e_W0 = (const float*)d_in[8];
  const float* enc_e_b0 = (const float*)d_in[9];
  const float* enc_e_W1 = (const float*)d_in[10];
  const float* enc_e_b1 = (const float*)d_in[11];
  const float* W_msg    = (const float*)d_in[12];
  const float* node_W0  = (const float*)d_in[13];
  const float* node_b0  = (const float*)d_in[14];
  const float* node_W1  = (const float*)d_in[15];
  const float* node_b1  = (const float*)d_in[16];
  const float* W_node   = (const float*)d_in[17];
  const float* ln_g     = (const float*)d_in[18];
  const float* ln_b     = (const float*)d_in[19];
  const float* dec_W0   = (const float*)d_in[20];
  const float* dec_b0   = (const float*)d_in[21];
  const float* dec_W1   = (const float*)d_in[22];
  const float* dec_b1   = (const float*)d_in[23];

  const int N = in_sizes[0] / 32;
  const int E = in_sizes[1] / 16;

  // workspace carve-up (aligned to 256B)
  char* p = (char*)d_ws;
  auto take = [&](size_t bytes) -> char* {
    char* q = p;
    p += (bytes + 255) & ~(size_t)255;
    return q;
  };
  // packed weights: ktiles*4096 halves each
  _Float16* pw_enc_n0 = (_Float16*)take((size_t)1 * 4096 * 2);
  _Float16* pw_enc_n1 = (_Float16*)take((size_t)4 * 4096 * 2);
  _Float16* pw_enc_e0 = (_Float16*)take((size_t)1 * 4096 * 2);
  _Float16* pw_enc_e1 = (_Float16*)take((size_t)4 * 4096 * 2);
  _Float16* pw_msg_s  = (_Float16*)take((size_t)4 * 4096 * 2);
  _Float16* pw_msg_e  = (_Float16*)take((size_t)4 * 4096 * 2);
  _Float16* pw_n0a    = (_Float16*)take((size_t)4 * 4096 * 2);
  _Float16* pw_n0b    = (_Float16*)take((size_t)4 * 4096 * 2);
  _Float16* pw_n1     = (_Float16*)take((size_t)4 * 4096 * 2);
  _Float16* pw_node   = (_Float16*)take((size_t)4 * 4096 * 2);
  _Float16* pw_d0     = (_Float16*)take((size_t)4 * 4096 * 2);
  _Float16* hw_d1     = (_Float16*)take((size_t)128 * 2 * 2);
  _Float16* e16       = (_Float16*)take((size_t)E * 128 * 2);  // encoded edges, f16
  float*    nA        = (float*)take((size_t)N * 128 * 4);
  float*    nB        = (float*)take((size_t)N * 128 * 4);
  float*    agg       = (float*)take((size_t)N * 128 * 4);

  auto pack = [&](_Float16* dst, const float* src, int ktiles, int K) {
    int total = ktiles * 4096;
    k_pack_b<<<(total + 255) / 256, 256, 0, stream>>>(dst, src, ktiles, K);
  };
  pack(pw_enc_n0, enc_n_W0, 1, 32);
  pack(pw_enc_n1, enc_n_W1, 4, 128);
  pack(pw_enc_e0, enc_e_W0, 1, 16);
  pack(pw_enc_e1, enc_e_W1, 4, 128);
  pack(pw_msg_s, W_msg, 4, 128);                 // rows 0..127   (sender part)
  pack(pw_msg_e, W_msg + 128 * HD, 4, 128);      // rows 128..255 (edge part)
  pack(pw_n0a, node_W0, 4, 128);                 // rows 0..127   (n part)
  pack(pw_n0b, node_W0 + 128 * HD, 4, 128);      // rows 128..255 (agg part)
  pack(pw_n1, node_W1, 4, 128);
  pack(pw_node, W_node, 4, 128);
  pack(pw_d0, dec_W0, 4, 128);
  k_f2h<<<1, 256, 0, stream>>>(hw_d1, dec_W1, 128 * 2);

  const int nblk = (N + 63) / 64;
  const int eblk = (E + 63) / 64;

  // encoders
  k_encode<<<nblk, 64, 0, stream>>>(nodes, 32, N, pw_enc_n0, enc_n_b0,
                                    pw_enc_n1, enc_n_b1, nA, nullptr);
  k_encode<<<eblk, 64, 0, stream>>>(edges, 16, E, pw_enc_e0, enc_e_b0,
                                    pw_enc_e1, enc_e_b1, nullptr, e16);

  // 5 weight-tied message-passing rounds
  float* cur = nA;
  float* nxt = nB;
  for (int pass = 0; pass < 5; ++pass) {
    hipMemsetAsync(agg, 0, (size_t)N * 128 * 4, stream);
    k_message<<<eblk, 64, 0, stream>>>(cur, e16, senders, receivers,
                                       pw_msg_s, pw_msg_e, agg, E);
    k_node_update<<<nblk, 64, 0, stream>>>(cur, agg, pw_n0a, pw_n0b, node_b0,
                                           pw_n1, node_b1, pw_node, nxt, N);
    k_layernorm<<<(N + 7) / 8, 256, 0, stream>>>(nxt, ln_g, ln_b, N);
    float* t = cur; cur = nxt; nxt = t;
  }

  // decoder
  k_decode<<<nblk, 64, 0, stream>>>(cur, pw_d0, dec_b0, hw_d1, dec_b1,
                                    (float*)d_out, N);
}
